// MultiHeadAttention_18287970746815
// MI455X (gfx1250) — matrix operand order repaired and checked by reading, not compiled, not run on hardware
//
#include <hip/hip_runtime.h>
#include <hip/hip_bf16.h>

typedef __attribute__((ext_vector_type(16))) _Float16 v16h;
typedef __attribute__((ext_vector_type(8)))  _Float16 v8h;
typedef __attribute__((ext_vector_type(8)))  float    v8f;
typedef __attribute__((ext_vector_type(4)))  float    v4f;

#define WMMA_F32_F16(a, b, c) \
  __builtin_amdgcn_wmma_f32_16x16x32_f16(false, (a), false, (b), (short)0, (c), false, false)

// A-matrix 16x32 f16 fragment, contiguous K order used here (not ISA table order): lane holds row M=lane%16,
// element e -> K = (lane/16)*16 + e.
// p = row_base + (lane/16)*16 : two contiguous 8-half chunks at +0 and +8.
__device__ __forceinline__ v16h frag_a_from(const _Float16* p) {
  v8h lo = *(const v8h*)(p);
  v8h hi = *(const v8h*)(p + 8);
  v16h r;
#pragma unroll
  for (int e = 0; e < 8; ++e) { r[e] = lo[e]; r[e + 8] = hi[e]; }
  return r;
}

// B-matrix 32x16 f16 fragment: lane holds col N=lane%16,
// element e -> K = (lane/16)*16 + e  (16 contiguous K values at p).
__device__ __forceinline__ v16h frag_b_from(const _Float16* p) {
  v8h lo = *(const v8h*)(p);
  v8h hi = *(const v8h*)(p + 8);
  v16h r;
#pragma unroll
  for (int e = 0; e < 8; ++e) { r[e] = lo[e]; r[e + 8] = hi[e]; }
  return r;
}

// ---------------------------------------------------------------------------
// Weight pre-pass: Wt[n][k] (f16) = (f16)W[k][n].  1024x1024, 32x32 LDS tiles.
// ---------------------------------------------------------------------------
__global__ __launch_bounds__(256) void w_transpose_f16(
    const float* __restrict__ W, _Float16* __restrict__ Wt) {
  __shared__ float tile[32][33];
  const int tx = threadIdx.x & 31, ty = threadIdx.x >> 5;  // 32 x 8
  const int nb = blockIdx.x * 32;   // n tile base
  const int kb = blockIdx.y * 32;   // k tile base
#pragma unroll
  for (int p = 0; p < 4; ++p) {
    int k = kb + ty + p * 8;
    tile[ty + p * 8][tx] = W[(size_t)k * 1024 + nb + tx];
  }
  __syncthreads();
#pragma unroll
  for (int p = 0; p < 4; ++p) {
    int n = nb + ty + p * 8;
    Wt[(size_t)n * 1024 + kb + tx] = (_Float16)tile[tx][ty + p * 8];
  }
}

// ---------------------------------------------------------------------------
// Projection GEMM: out[b,h,t,d] (f16) = X[m=(b,t), :] @ W[:, n=(h,d)] + bias[n]
// X fp32 [8192,1024], Wt f16 column-major [n][k]. 256 thr (8 waves),
// block tile 128x128; wave w owns M-strip w*16 and all 128 N cols.
// Double-buffered LDS + 2-deep B-fragment pipeline (partial dscnt waits).
// ---------------------------------------------------------------------------
__global__ __launch_bounds__(256) void mha_proj_gemm(
    const float* __restrict__ X, const _Float16* __restrict__ Wt,
    const float* __restrict__ bias, _Float16* __restrict__ out) {
  __shared__ __align__(16) _Float16 Ash[2][128 * 40];   // [128 m][32 k + pad]
  __shared__ __align__(16) _Float16 Bsh[2][128 * 40];   // [128 n][32 k + pad]
  const int tid  = threadIdx.x;
  const int lane = tid & 31, wave = tid >> 5;
  const int n15  = lane & 15, half = lane >> 4;
  const int m0 = blockIdx.y * 128;
  const int n0 = blockIdx.x * 128;

  const int srow = tid >> 1;              // 0..127
  const int scol = (tid & 1) * 16;        // 0 or 16
  const float*    xrow = &X[(size_t)(m0 + srow) * 1024 + scol];
  const _Float16* wrow = &Wt[(size_t)(n0 + srow) * 1024 + scol];

  v4f x0, x1, x2, x3;
  v8h wb0, wb1;

  auto load_regs = [&](int k0) {
    x0 = *(const v4f*)(xrow + k0 + 0);
    x1 = *(const v4f*)(xrow + k0 + 4);
    x2 = *(const v4f*)(xrow + k0 + 8);
    x3 = *(const v4f*)(xrow + k0 + 12);
    wb0 = *(const v8h*)(wrow + k0);
    wb1 = *(const v8h*)(wrow + k0 + 8);
  };
  auto store_regs = [&](int buf) {
    v8h h0, h1;
#pragma unroll
    for (int e = 0; e < 4; ++e) {
      h0[e] = (_Float16)x0[e]; h0[e + 4] = (_Float16)x1[e];
      h1[e] = (_Float16)x2[e]; h1[e + 4] = (_Float16)x3[e];
    }
    *(v8h*)&Ash[buf][srow * 40 + scol]     = h0;
    *(v8h*)&Ash[buf][srow * 40 + scol + 8] = h1;
    *(v8h*)&Bsh[buf][srow * 40 + scol]     = wb0;
    *(v8h*)&Bsh[buf][srow * 40 + scol + 8] = wb1;
  };

  v8f acc[8] = {};

  load_regs(0);
  store_regs(0);
  __syncthreads();

  for (int step = 0; step < 32; ++step) {
    const int cur = step & 1;
    if (step < 31) load_regs((step + 1) * 32);   // issue next tile's globals now

    v16h a  = frag_a_from(&Ash[cur][(wave * 16 + n15) * 40 + half * 16]);
    // 2-deep B fragment pipeline: frags f+1 and f+2 stay in flight
    v16h b0 = frag_b_from(&Bsh[cur][n15 * 40 + half * 16]);
    v16h b1 = frag_b_from(&Bsh[cur][(16 + n15) * 40 + half * 16]);
#pragma unroll
    for (int f = 0; f < 8; ++f) {
      v16h curb = b0;
      b0 = b1;
      if (f < 6) b1 = frag_b_from(&Bsh[cur][((f + 2) * 16 + n15) * 40 + half * 16]);
      acc[f] = WMMA_F32_F16(a, curb, acc[f]);
    }

    if (step < 31) store_regs(cur ^ 1);
    __syncthreads();
  }

#pragma unroll
  for (int f = 0; f < 8; ++f) {
    const int n = n0 + f * 16 + n15;
    const float bn = bias[n];
    const int hh = n >> 6, dd = n & 63;
#pragma unroll
    for (int r = 0; r < 8; ++r) {
      int m = m0 + wave * 16 + r + 8 * half;
      int bb = m >> 10, t = m & 1023;
      out[(((size_t)bb * 16 + hh) * 1024 + t) * 64 + dd] = (_Float16)(acc[f][r] + bn);
    }
  }
}

// ---------------------------------------------------------------------------
// Output GEMM: Y[m,n] (fp32) = Ao[(b,h,t,d)] @ Wo + bo. Ao f16 head-major,
// Wto f16 column-major. Same 128x128 double-buffered tiling.
// ---------------------------------------------------------------------------
__global__ __launch_bounds__(256) void mha_out_gemm(
    const _Float16* __restrict__ A, const _Float16* __restrict__ Wt,
    const float* __restrict__ bias, float* __restrict__ Y) {
  __shared__ __align__(16) _Float16 Ash[2][128 * 40];
  __shared__ __align__(16) _Float16 Bsh[2][128 * 40];
  const int tid  = threadIdx.x;
  const int lane = tid & 31, wave = tid >> 5;
  const int n15  = lane & 15, half = lane >> 4;
  const int m0 = blockIdx.y * 128;
  const int n0 = blockIdx.x * 128;

  const int srow = tid >> 1;
  const int scol = (tid & 1) * 16;
  // A row base in head-major layout: m -> (b,t); k-chunk selects head hh = k>>6
  const int m = m0 + srow;
  const int bb = m >> 10, t = m & 1023;
  const _Float16* arow = &A[((size_t)bb * 16 * 1024 + t) * 64];  // + hh*1024*64 + dd
  const _Float16* wrow = &Wt[(size_t)(n0 + srow) * 1024 + scol];

  v8h ab0, ab1, wb0, wb1;

  auto load_regs = [&](int k0) {
    int kk = k0 + scol;
    int hh = kk >> 6, dd = kk & 63;
    const _Float16* ap = arow + (size_t)hh * 1024 * 64 + dd;
    ab0 = *(const v8h*)(ap);
    ab1 = *(const v8h*)(ap + 8);
    wb0 = *(const v8h*)(wrow + k0);
    wb1 = *(const v8h*)(wrow + k0 + 8);
  };
  auto store_regs = [&](int buf) {
    *(v8h*)&Ash[buf][srow * 40 + scol]     = ab0;
    *(v8h*)&Ash[buf][srow * 40 + scol + 8] = ab1;
    *(v8h*)&Bsh[buf][srow * 40 + scol]     = wb0;
    *(v8h*)&Bsh[buf][srow * 40 + scol + 8] = wb1;
  };

  v8f acc[8] = {};

  load_regs(0);
  store_regs(0);
  __syncthreads();

  for (int step = 0; step < 32; ++step) {
    const int cur = step & 1;
    if (step < 31) load_regs((step + 1) * 32);

    v16h a  = frag_a_from(&Ash[cur][(wave * 16 + n15) * 40 + half * 16]);
    v16h b0 = frag_b_from(&Bsh[cur][n15 * 40 + half * 16]);
    v16h b1 = frag_b_from(&Bsh[cur][(16 + n15) * 40 + half * 16]);
#pragma unroll
    for (int f = 0; f < 8; ++f) {
      v16h curb = b0;
      b0 = b1;
      if (f < 6) b1 = frag_b_from(&Bsh[cur][((f + 2) * 16 + n15) * 40 + half * 16]);
      acc[f] = WMMA_F32_F16(a, curb, acc[f]);
    }

    if (step < 31) store_regs(cur ^ 1);
    __syncthreads();
  }

#pragma unroll
  for (int f = 0; f < 8; ++f) {
    const int n = n0 + f * 16 + n15;
    const float bn = bias[n];
#pragma unroll
    for (int r = 0; r < 8; ++r) {
      int mm = m0 + wave * 16 + r + 8 * half;
      Y[(size_t)mm * 1024 + n] = acc[f][r] + bn;
    }
  }
}

// ---------------------------------------------------------------------------
// Flash attention: one block = one (b,h) x 64 query rows. 4 waves, each wave
// owns 16 query rows. Key tiles of 32 with online softmax.
// Faithful masking: score = (mask==0) ? -1e-9 : score*scale.
// ---------------------------------------------------------------------------
__global__ __launch_bounds__(128) void mha_attention(
    const _Float16* __restrict__ Q, const _Float16* __restrict__ K,
    const _Float16* __restrict__ V, const int* __restrict__ mask,
    _Float16* __restrict__ O) {
  __shared__ __align__(16) _Float16 Ksh[32 * 72];     // [32 key][64 d + pad]
  __shared__ __align__(16) _Float16 Vt[64 * 40];      // [64 d][32 key + pad]
  __shared__ __align__(16) _Float16 Psh[4 * 16 * 40]; // per-wave P [16][32 + pad]
  __shared__ __align__(16) int      Msh[64 * 36];     // [64 t][32 s + pad]

  const int tid  = threadIdx.x;
  const int lane = tid & 31, wave = tid >> 5;
  const int n15  = lane & 15, half = lane >> 4;
  const int bh = blockIdx.y;          // b*16 + h
  const int b  = bh >> 4;
  const int q0 = blockIdx.x * 64;
  const int qrow = q0 + wave * 16;

  const _Float16* qp = Q + ((size_t)bh * 1024 + (qrow + n15)) * 64;
  const v16h Qa0 = frag_a_from(qp + half * 16);
  const v16h Qa1 = frag_a_from(qp + 32 + half * 16);

  const _Float16* kp = K + (size_t)bh * 1024 * 64;
  const _Float16* vp = V + (size_t)bh * 1024 * 64;
  const size_t mbase0 = ((size_t)b * 1024 + q0) * 1024;

  float mrun[8], lrun[8];
  v8f Oacc[4] = {};
#pragma unroll
  for (int r = 0; r < 8; ++r) { mrun[r] = -3.0e38f; lrun[r] = 0.0f; }

  const float scale = 0.125f;  // 1/sqrt(64)

  for (int kt = 0; kt < 1024; kt += 32) {
    __syncthreads();
    // Stage K (row-major), V (transposed), mask tile (int4 coalesced)
#pragma unroll
    for (int p = 0; p < 2; ++p) {
      int c = tid + p * 128;                 // 0..255
      int keyr = c >> 3, d8 = (c & 7) * 8;
      *(v8h*)&Ksh[keyr * 72 + d8] = *(const v8h*)&kp[(size_t)(kt + keyr) * 64 + d8];
      v8h vv = *(const v8h*)&vp[(size_t)(kt + keyr) * 64 + d8];
#pragma unroll
      for (int e = 0; e < 8; ++e) Vt[(d8 + e) * 40 + keyr] = vv[e];
    }
#pragma unroll
    for (int p = 0; p < 4; ++p) {
      int c = tid + p * 128;                 // 0..511
      int rowm = c >> 3, col4 = (c & 7) * 4;
      *(int4*)&Msh[rowm * 36 + col4] =
          *(const int4*)&mask[mbase0 + (size_t)rowm * 1024 + kt + col4];
    }
    // prefetch next K/V tile (global_prefetch_b8)
    if (kt + 32 < 1024) {
      __builtin_prefetch(kp + (size_t)(kt + 32) * 64 + tid * 16, 0, 1);
      __builtin_prefetch(vp + (size_t)(kt + 32) * 64 + tid * 16, 0, 1);
    }
    __syncthreads();

    // S = Q @ K^T for two 16-key subtiles (d = 64 -> 2 WMMAs each)
    v8f s[2];
#pragma unroll
    for (int j = 0; j < 2; ++j) {
      const _Float16* kr = &Ksh[(j * 16 + n15) * 72];
      v16h kb0 = frag_b_from(kr + half * 16);
      v16h kb1 = frag_b_from(kr + 32 + half * 16);
      v8f c = {};
      c = WMMA_F32_F16(Qa0, kb0, c);
      c = WMMA_F32_F16(Qa1, kb1, c);
      s[j] = c;
    }

    // scale + mask (mask==0 -> -1e-9, faithful to reference)
#pragma unroll
    for (int j = 0; j < 2; ++j)
#pragma unroll
      for (int r = 0; r < 8; ++r) {
        int mv = Msh[(wave * 16 + r + 8 * half) * 36 + j * 16 + n15];
        float val = s[j][r] * scale;
        s[j][r] = (mv == 0) ? -1e-9f : val;
      }

    // online softmax; row stats shared across the 16-lane group
#pragma unroll
    for (int r = 0; r < 8; ++r) {
      float mt = fmaxf(s[0][r], s[1][r]);
#pragma unroll
      for (int off = 1; off < 16; off <<= 1) mt = fmaxf(mt, __shfl_xor(mt, off, 16));
      float mnew  = fmaxf(mrun[r], mt);
      float alpha = __expf(mrun[r] - mnew);
      float p0 = __expf(s[0][r] - mnew);
      float p1 = __expf(s[1][r] - mnew);
      float rs = p0 + p1;
#pragma unroll
      for (int off = 1; off < 16; off <<= 1) rs += __shfl_xor(rs, off, 16);
      lrun[r] = lrun[r] * alpha + rs;
      mrun[r] = mnew;
      s[0][r] = p0; s[1][r] = p1;
#pragma unroll
      for (int f = 0; f < 4; ++f) Oacc[f][r] *= alpha;
    }

    // P (C-layout) -> per-wave LDS -> A-layout fragment
    _Float16* Pw = &Psh[wave * 16 * 40];
#pragma unroll
    for (int j = 0; j < 2; ++j)
#pragma unroll
      for (int r = 0; r < 8; ++r)
        Pw[(r + 8 * half) * 40 + j * 16 + n15] = (_Float16)s[j][r];
    __syncthreads();

    v16h pa = frag_a_from(&Pw[n15 * 40 + half * 16]);

    // O += P @ V : 4 d-tiles; 2-deep fragment pipeline from transposed Vt
    v16h vb0 = frag_b_from(&Vt[n15 * 40 + half * 16]);
    v16h vb1 = frag_b_from(&Vt[(16 + n15) * 40 + half * 16]);
#pragma unroll
    for (int f = 0; f < 4; ++f) {
      v16h curb = vb0;
      vb0 = vb1;
      if (f < 2) vb1 = frag_b_from(&Vt[((f + 2) * 16 + n15) * 40 + half * 16]);
      Oacc[f] = WMMA_F32_F16(pa, curb, Oacc[f]);
    }
  }

  // normalize and store f16 [B,H,T,D]
  _Float16* op = O + (size_t)bh * 1024 * 64;
#pragma unroll
  for (int r = 0; r < 8; ++r) {
    const float inv = 1.0f / lrun[r];
    const int t = qrow + r + 8 * half;
#pragma unroll
    for (int f = 0; f < 4; ++f) {
      int d = f * 16 + n15;
      op[(size_t)t * 64 + d] = (_Float16)(Oacc[f][r] * inv);
    }
  }
}

// ---------------------------------------------------------------------------
extern "C" void kernel_launch(void* const* d_in, const int* in_sizes, int n_in,
                              void* d_out, int out_size, void* d_ws, size_t ws_size,
                              hipStream_t stream) {
  const float* query = (const float*)d_in[0];
  const float* key   = (const float*)d_in[1];
  const float* value = (const float*)d_in[2];
  const int*   mask  = (const int*)d_in[3];
  const float* Wq = (const float*)d_in[4];
  const float* bq = (const float*)d_in[5];
  const float* Wk = (const float*)d_in[6];
  const float* bk = (const float*)d_in[7];
  const float* Wv = (const float*)d_in[8];
  const float* bv = (const float*)d_in[9];
  const float* Wo = (const float*)d_in[10];
  const float* bo = (const float*)d_in[11];

  const size_t NELT = (size_t)8 * 1024 * 1024;   // B*T*C
  const size_t WELT = (size_t)1024 * 1024;
  _Float16* qws = (_Float16*)d_ws;
  _Float16* kws = qws + NELT;
  _Float16* vws = kws + NELT;
  _Float16* aws = vws + NELT;
  _Float16* wtq = aws + NELT;
  _Float16* wtk = wtq + WELT;
  _Float16* wtv = wtk + WELT;
  _Float16* wto = wtv + WELT;

  dim3 gT(32, 32);
  w_transpose_f16<<<gT, 256, 0, stream>>>(Wq, wtq);
  w_transpose_f16<<<gT, 256, 0, stream>>>(Wk, wtk);
  w_transpose_f16<<<gT, 256, 0, stream>>>(Wv, wtv);
  w_transpose_f16<<<gT, 256, 0, stream>>>(Wo, wto);

  dim3 gProj(8, 64);                             // N tiles x M tiles (128x128)
  mha_proj_gemm<<<gProj, 256, 0, stream>>>(query, wtq, bq, qws);
  mha_proj_gemm<<<gProj, 256, 0, stream>>>(key,   wtk, bk, kws);
  mha_proj_gemm<<<gProj, 256, 0, stream>>>(value, wtv, bv, vws);
  mha_attention<<<dim3(16, 128), 128, 0, stream>>>(qws, kws, vws, mask, aws);
  mha_out_gemm<<<gProj, 256, 0, stream>>>(aws, wto, bo, (float*)d_out);

  (void)in_sizes; (void)n_in; (void)out_size; (void)ws_size;
}
